// AttentionMechanism_36936718746027
// MI455X (gfx1250) — compile-verified, matrix operand order checked by reading, not run on hardware
//
#include <hip/hip_runtime.h>
#include <hip/hip_bf16.h>

typedef __bf16 bf16_t;
typedef __attribute__((ext_vector_type(16))) __bf16 v16bf;
typedef __attribute__((ext_vector_type(8)))  __bf16 v8bf;
typedef __attribute__((ext_vector_type(4)))  __bf16 v4bf;
typedef __attribute__((ext_vector_type(8)))  float  v8f;

#define D_MODEL 1024
#define N_HEADS 16
#define DK      64
#define BATCH   2
#define SEQ     2048
#define ROWS    (BATCH*SEQ)   /* 4096 */

// round-to-nearest-even f32 -> bf16
__device__ __forceinline__ bf16_t f2bf(float f) {
  unsigned u = __float_as_uint(f);
  unsigned r = (u + 0x7FFFu + ((u >> 16) & 1u)) >> 16;
  unsigned short s = (unsigned short)r;
  return __builtin_bit_cast(bf16_t, s);
}

// Per-lane 16-bf16 fragment of a 16x16x32 WMMA A/B operand.
// base points at (row-or-col)*ld + kbase; lane needs k = 8*half + [0..7]
// and k = 16 + 8*half + [0..7]  (two 16-byte contiguous chunks).
__device__ __forceinline__ v16bf load_frag(const bf16_t* base, int half) {
  v8bf lo = *reinterpret_cast<const v8bf*>(base + 8 * half);
  v8bf hi = *reinterpret_cast<const v8bf*>(base + 16 + 8 * half);
  return __builtin_shufflevector(lo, hi, 0,1,2,3,4,5,6,7,8,9,10,11,12,13,14,15);
}

__device__ __forceinline__ v8f wmma_bf16(v16bf a, v16bf b, v8f c) {
  return __builtin_amdgcn_wmma_f32_16x16x32_bf16(false, a, false, b, (short)0, c, false, false);
}

// ---------------- elementwise f32 -> bf16 convert (for x) ----------------
__global__ __launch_bounds__(256) void cvt_kernel(const float* __restrict__ in,
                                                  bf16_t* __restrict__ out, int n4) {
  int i = blockIdx.x * blockDim.x + threadIdx.x;
  if (i < n4) {
    float4 v = reinterpret_cast<const float4*>(in)[i];
    v4bf o;
    o[0] = f2bf(v.x); o[1] = f2bf(v.y); o[2] = f2bf(v.z); o[3] = f2bf(v.w);
    reinterpret_cast<v4bf*>(out)[i] = o;
  }
}

// ------------- transpose + convert: W(f32, KxN) -> Wt(bf16, NxK) -------------
__global__ __launch_bounds__(256) void transpose_cvt_kernel(const float* __restrict__ W,
                                                            bf16_t* __restrict__ Wt) {
  __shared__ float tile[64][65];
  int tid = threadIdx.x;
  int r = tid >> 6;      // 0..3
  int c = tid & 63;      // 0..63
  int kbase = blockIdx.y * 64;
  int nbase = blockIdx.x * 64;
#pragma unroll
  for (int i = 0; i < 16; ++i)
    tile[r + 4 * i][c] = W[(size_t)(kbase + r + 4 * i) * D_MODEL + nbase + c];
  __syncthreads();
#pragma unroll
  for (int i = 0; i < 16; ++i)
    Wt[(size_t)(nbase + r + 4 * i) * D_MODEL + kbase + c] = f2bf(tile[c][r + 4 * i]);
}

// ---------------- GEMM: C(MxN) = A(MxK, bf16) * Bt(NxK, bf16)^T ----------------
// Each wave: 32(M) x 64(N) tile -> 8 WMMAs per k-step, B-frags reused across
// the two 16-row M-subtiles (~22 FLOP/byte from cache).
// mode 0: bf16 out, per-head (b,h,s,d) layout (Q/K)
// mode 1: bf16 out, per-head transposed (b,h,d,s) layout (V^T)
// mode 2: f32 out, row-major (final projection)
__global__ __launch_bounds__(256) void gemm_kernel(const bf16_t* __restrict__ A,
                                                   const bf16_t* __restrict__ Bt,
                                                   bf16_t* __restrict__ outb,
                                                   float* __restrict__ outf,
                                                   int N, int Kd, int mode) {
  int tid  = threadIdx.x;
  int wid  = tid >> 5;
  int lane = tid & 31;
  int n16  = lane & 15;
  int half = lane >> 4;
  int rowBase = blockIdx.y * 64 + (wid >> 2) * 32;    // 2 waves along M (32 rows each)
  int colBase = blockIdx.x * 256 + (wid & 3) * 64;    // 4 waves along N (64 cols each)

  v8f acc0[4] = {}, acc1[4] = {};
  const bf16_t* aRow0 = A  + (size_t)(rowBase + n16) * Kd;
  const bf16_t* aRow1 = aRow0 + (size_t)16 * Kd;
  const bf16_t* bRow  = Bt + (size_t)(colBase + n16) * Kd;

  for (int kb = 0; kb < Kd; kb += 32) {
    v16bf a0 = load_frag(aRow0 + kb, half);
    v16bf a1 = load_frag(aRow1 + kb, half);
    __builtin_prefetch(aRow0 + kb + 128, 0, 1);       // global_prefetch next tiles
    __builtin_prefetch(aRow1 + kb + 128, 0, 1);
#pragma unroll
    for (int c = 0; c < 4; ++c) {
      v16bf bf = load_frag(bRow + (size_t)(c * 16) * Kd + kb, half);
      acc0[c] = wmma_bf16(a0, bf, acc0[c]);
      acc1[c] = wmma_bf16(a1, bf, acc1[c]);
    }
  }

#pragma unroll
  for (int t = 0; t < 2; ++t) {
#pragma unroll
    for (int c = 0; c < 4; ++c) {
#pragma unroll
      for (int r = 0; r < 8; ++r) {
        int row = rowBase + t * 16 + r + 8 * half;    // C-layout row
        int col = colBase + c * 16 + n16;
        float v = (t == 0) ? acc0[c][r] : acc1[c][r];
        if (mode == 2) {
          outf[(size_t)row * N + col] = v;
        } else if (mode == 0) {
          int b = row >> 11, s = row & 2047;
          int h = col >> 6,  d = col & 63;
          outb[(((size_t)(b * N_HEADS + h)) * SEQ + s) * DK + d] = f2bf(v);
        } else {
          int b = row >> 11, s = row & 2047;
          outb[((size_t)(b * D_MODEL + col)) * SEQ + s] = f2bf(v);
        }
      }
    }
  }
}

// ------------- flash attention: 32 queries/wave, 32 keys/iter ----------------
// K- and V-fragments are loaded once per iteration and reused by both 16-query
// subtiles: 16 WMMAs per 8 KB of K/V traffic.
__global__ __launch_bounds__(256) void attn_kernel(const bf16_t* __restrict__ Q,
                                                   const bf16_t* __restrict__ Kb,
                                                   const bf16_t* __restrict__ Vt,
                                                   bf16_t* __restrict__ Ob) {
  __shared__ __align__(16) bf16_t Plds[8][2 * 16 * 32];   // 2 KB per wave
  int tid  = threadIdx.x;
  int wid  = tid >> 5;
  int lane = tid & 31;
  int n16  = lane & 15;
  int half = lane >> 4;

  int unit = blockIdx.x * 8 + wid;    // one (bh, 32-query block) per wave
  int bh   = unit >> 6;               // / (SEQ/32)
  int q0   = (unit & 63) * 32;
  int b    = bh >> 4;
  int h    = bh & 15;

  const bf16_t* Qp = Q  + (size_t)bh * SEQ * DK;
  const bf16_t* Kp = Kb + (size_t)bh * SEQ * DK;
  const bf16_t* Vp = Vt + (size_t)bh * DK * SEQ;

  v16bf qf[2][2];
#pragma unroll
  for (int t = 0; t < 2; ++t) {
    const bf16_t* qp = Qp + (size_t)(q0 + t * 16 + n16) * DK;
    qf[t][0] = load_frag(qp, half);
    qf[t][1] = load_frag(qp + 32, half);
  }

  v8f o[2][4] = {};
  float mi[2][8], li[2][8];
#pragma unroll
  for (int t = 0; t < 2; ++t)
#pragma unroll
    for (int r = 0; r < 8; ++r) { mi[t][r] = -__builtin_inff(); li[t][r] = 0.f; }

  bf16_t* myP = &Plds[wid][0];
  const float scale = 0.125f;   // 1/sqrt(64)

  for (int j0 = 0; j0 < SEQ; j0 += 32) {
    // K fragments, loaded once for both query subtiles
    v16bf kf[2][2];
#pragma unroll
    for (int sub = 0; sub < 2; ++sub) {
      const bf16_t* kp = Kp + (size_t)(j0 + sub * 16 + n16) * DK;
      kf[sub][0] = load_frag(kp, half);
      kf[sub][1] = load_frag(kp + 32, half);
    }

#pragma unroll
    for (int t = 0; t < 2; ++t) {
      v8f s0 = {}, s1 = {};
      s0 = wmma_bf16(qf[t][0], kf[0][0], s0);
      s0 = wmma_bf16(qf[t][1], kf[0][1], s0);
      s1 = wmma_bf16(qf[t][0], kf[1][0], s1);
      s1 = wmma_bf16(qf[t][1], kf[1][1], s1);

      float rmax[8];
#pragma unroll
      for (int r = 0; r < 8; ++r) {
        float a = s0[r] * scale; s0[r] = a;
        float bb = s1[r] * scale; s1[r] = bb;
        float v = fmaxf(a, bb);               // row max over 32 keys
        v = fmaxf(v, __shfl_xor(v, 1, 32));
        v = fmaxf(v, __shfl_xor(v, 2, 32));
        v = fmaxf(v, __shfl_xor(v, 4, 32));
        v = fmaxf(v, __shfl_xor(v, 8, 32));
        rmax[r] = v;
      }

      bf16_t* pT = myP + t * 512;
#pragma unroll
      for (int r = 0; r < 8; ++r) {
        float mn   = fmaxf(mi[t][r], rmax[r]);
        float corr = __expf(mi[t][r] - mn);
        mi[t][r] = mn;
        float p0 = __expf(s0[r] - mn);
        float p1 = __expf(s1[r] - mn);
        int row = r + 8 * half;               // C-layout row -> LDS row
        pT[row * 32 + n16]      = f2bf(p0);
        pT[row * 32 + 16 + n16] = f2bf(p1);
        float ps = p0 + p1;
        ps += __shfl_xor(ps, 1, 32);
        ps += __shfl_xor(ps, 2, 32);
        ps += __shfl_xor(ps, 4, 32);
        ps += __shfl_xor(ps, 8, 32);
        li[t][r] = li[t][r] * corr + ps;
#pragma unroll
        for (int c = 0; c < 4; ++c) o[t][c][r] *= corr;
      }
    }
    __syncthreads();

    // V fragments loaded once, reused by both query subtiles
    v16bf vf[4];
#pragma unroll
    for (int c = 0; c < 4; ++c)
      vf[c] = load_frag(Vp + (size_t)(c * 16 + n16) * SEQ + j0, half);

#pragma unroll
    for (int t = 0; t < 2; ++t) {
      v16bf pf = load_frag(myP + t * 512 + n16 * 32, half);
#pragma unroll
      for (int c = 0; c < 4; ++c)
        o[t][c] = wmma_bf16(pf, vf[c], o[t][c]);
    }
    __syncthreads();
  }

  // normalize + write concat (b, s, h*64+d) as bf16
#pragma unroll
  for (int t = 0; t < 2; ++t) {
#pragma unroll
    for (int r = 0; r < 8; ++r) {
      float inv = 1.0f / li[t][r];
      int rowG = b * SEQ + q0 + t * 16 + r + 8 * half;
      size_t base = (size_t)rowG * D_MODEL + h * DK + n16;
      Ob[base + 0]  = f2bf(o[t][0][r] * inv);
      Ob[base + 16] = f2bf(o[t][1][r] * inv);
      Ob[base + 32] = f2bf(o[t][2][r] * inv);
      Ob[base + 48] = f2bf(o[t][3][r] * inv);
    }
  }
}

extern "C" void kernel_launch(void* const* d_in, const int* in_sizes, int n_in,
                              void* d_out, int out_size, void* d_ws, size_t ws_size,
                              hipStream_t stream) {
  const float* x  = (const float*)d_in[0];
  const float* Wq = (const float*)d_in[1];
  const float* Wk = (const float*)d_in[2];
  const float* Wv = (const float*)d_in[3];
  const float* Wo = (const float*)d_in[4];
  float* out = (float*)d_out;

  char* ws = (char*)d_ws;
  size_t off = 0;
  const size_t XB = (size_t)ROWS * D_MODEL * sizeof(bf16_t);       // 8 MB
  const size_t WB = (size_t)D_MODEL * D_MODEL * sizeof(bf16_t);    // 2 MB
  bf16_t* xb  = (bf16_t*)(ws + off); off += XB;
  bf16_t* Wtq = (bf16_t*)(ws + off); off += WB;
  bf16_t* Wtk = (bf16_t*)(ws + off); off += WB;
  bf16_t* Wtv = (bf16_t*)(ws + off); off += WB;
  bf16_t* Wto = (bf16_t*)(ws + off); off += WB;
  bf16_t* Qb  = (bf16_t*)(ws + off); off += XB;
  bf16_t* Kb  = (bf16_t*)(ws + off); off += XB;
  bf16_t* Vtb = (bf16_t*)(ws + off); off += XB;
  bf16_t* Ob  = (bf16_t*)(ws + off); off += XB;   // total ~48 MB

  // 1) convert x to bf16
  int n4 = ROWS * D_MODEL / 4;
  cvt_kernel<<<(n4 + 255) / 256, 256, 0, stream>>>(x, xb, n4);

  // 2) transpose+convert weights (Wt[n][k] = W[k][n])
  dim3 tg(D_MODEL / 64, D_MODEL / 64);
  transpose_cvt_kernel<<<tg, 256, 0, stream>>>(Wq, Wtq);
  transpose_cvt_kernel<<<tg, 256, 0, stream>>>(Wk, Wtk);
  transpose_cvt_kernel<<<tg, 256, 0, stream>>>(Wv, Wtv);
  transpose_cvt_kernel<<<tg, 256, 0, stream>>>(Wo, Wto);

  // 3) projections (WMMA bf16): block tile 64(M) x 256(N), wave tile 32x64
  dim3 gg(D_MODEL / 256, ROWS / 64);
  gemm_kernel<<<gg, 256, 0, stream>>>(xb, Wtq, Qb,  nullptr, D_MODEL, D_MODEL, 0);
  gemm_kernel<<<gg, 256, 0, stream>>>(xb, Wtk, Kb,  nullptr, D_MODEL, D_MODEL, 0);
  gemm_kernel<<<gg, 256, 0, stream>>>(xb, Wtv, Vtb, nullptr, D_MODEL, D_MODEL, 1);

  // 4) flash attention: 2048 (bh, q32) units, 8 waves/block
  attn_kernel<<<(BATCH * N_HEADS * (SEQ / 32)) / 8, 256, 0, stream>>>(Qb, Kb, Vtb, Ob);

  // 5) output projection, f32 result
  gemm_kernel<<<gg, 256, 0, stream>>>(Ob, Wto, nullptr, out, D_MODEL, D_MODEL, 2);
}
